// MultiheadAttention_83751862272390
// MI455X (gfx1250) — matrix operand order repaired and checked by reading, not compiled, not run on hardware
//
#include <hip/hip_runtime.h>

typedef __attribute__((ext_vector_type(4)))  float        f4;
typedef __attribute__((ext_vector_type(8)))  float        v8f;
typedef __attribute__((ext_vector_type(16))) __bf16       v16bf;
typedef __attribute__((ext_vector_type(4)))  __bf16       bf4;
typedef __attribute__((ext_vector_type(4)))  unsigned int u4;
typedef int gi4 __attribute__((vector_size(16)));   // matches builtin param type

#define AS1 __attribute__((address_space(1)))
#define AS3 __attribute__((address_space(3)))

#if __has_builtin(__builtin_amdgcn_global_load_async_to_lds_b128) && \
    __has_builtin(__builtin_amdgcn_s_wait_asynccnt)
#define HAVE_ASYNC_LDS 1
#else
#define HAVE_ASYNC_LDS 0
#endif

#define Bsz 32
#define Ssz 512
#define Esz 512
#define NKV 2048   // packed kv columns: [kw | vw | kE | vE], 512 each
#define Mtot 16384 // B*S rows

// Map packed kv-column n (0..2047) to W_qkv row index j.
__device__ __forceinline__ int jmap(int n) { return (n < 1024) ? (512 + n) : (1024 + n); }

// ---------------------------------------------------------------------------
// Kernel 0a: split X (fp32) into bf16 hi/lo pair (fp32 emulation operands)
// ---------------------------------------------------------------------------
__global__ __launch_bounds__(256) void convert_x(
    const float* __restrict__ X, __bf16* __restrict__ hi, __bf16* __restrict__ lo)
{
  size_t i = ((size_t)blockIdx.x * 256 + threadIdx.x) * 4;
  f4 x = *(const f4*)(X + i);
  bf4 h, l;
#pragma unroll
  for (int c = 0; c < 4; ++c) {
    __bf16 hh = (__bf16)x[c];
    h[c] = hh;
    l[c] = (__bf16)(x[c] - (float)hh);
  }
  *(bf4*)(hi + i) = h;
  *(bf4*)(lo + i) = l;
}

// ---------------------------------------------------------------------------
// Kernel 0b: gather the 2048 needed W_qkv rows (kw,vw,kE,vE) packed, split hi/lo
// ---------------------------------------------------------------------------
__global__ __launch_bounds__(256) void convert_w(
    const float* __restrict__ W, __bf16* __restrict__ hi, __bf16* __restrict__ lo)
{
  size_t i = ((size_t)blockIdx.x * 256 + threadIdx.x) * 4;   // over 2048*512
  int n   = (int)(i >> 9);
  int col = (int)(i & 511);
  f4 x = *(const f4*)(W + (size_t)jmap(n) * 512 + col);
  bf4 h, l;
#pragma unroll
  for (int c = 0; c < 4; ++c) {
    __bf16 hh = (__bf16)x[c];
    h[c] = hh;
    l[c] = (__bf16)(x[c] - (float)hh);
  }
  *(bf4*)(hi + i) = h;
  *(bf4*)(lo + i) = l;
}

// ---------------------------------------------------------------------------
// Fragment loaders (LDS, 16-bit layouts per ISA 7.12.2)
// A 16x32: lane(m,half): elems 0-7 = K[8h..8h+7], elems 8-15 = K[16+8h..+7]
// B 32x16: lane(n,half): elems 0-7 = K[8h..8h+7], elems 8-15 = K[16+8h..+7]
// ---------------------------------------------------------------------------
__device__ __forceinline__ v16bf load_a_frag(const __bf16* rowp, int half) {
  union { u4 u[2]; v16bf v; } f;
  f.u[0] = *(const u4*)(rowp + 8 * half);
  f.u[1] = *(const u4*)(rowp + 16 + 8 * half);
  return f.v;
}
__device__ __forceinline__ v16bf load_b_frag(const __bf16* rowp, int half) {
  union { u4 u[2]; v16bf v; } f;
  f.u[0] = *(const u4*)(rowp + 8 * half);
  f.u[1] = *(const u4*)(rowp + 16 + 8 * half);
  return f.v;
}

// ---------------------------------------------------------------------------
// Kernel 1: KV[16384,2048] = X @ Wsel^T + bias, bf16x3 fp32-emulation WMMA.
// Block tile 128x128, 8 waves, each wave owns 2(M) x 4(N) 16x16 subtiles.
// Operand staging uses GLOBAL_LOAD_ASYNC_TO_LDS_B128 (ASYNCcnt) when available.
// ---------------------------------------------------------------------------
#define LD 40   // LDS row stride in bf16 (80 B: conflict-free, 16B aligned)
__global__ __launch_bounds__(256) void qkv_kv_gemm(
    const __bf16* __restrict__ Xhi, const __bf16* __restrict__ Xlo,
    const __bf16* __restrict__ Whi, const __bf16* __restrict__ Wlo,
    const float* __restrict__ bqkv, float* __restrict__ KV)
{
  __shared__ __bf16 AhiS[128 * LD];
  __shared__ __bf16 AloS[128 * LD];
  __shared__ __bf16 BhiS[128 * LD];
  __shared__ __bf16 BloS[128 * LD];

  const int t = threadIdx.x;
  const int rowBase = blockIdx.y * 128;
  const int colBase = blockIdx.x * 128;
  const int wave = t >> 5;
  const int lane = t & 31;
  const int sub16 = lane & 15;
  const int half  = lane >> 4;
  const int waveM = wave & 3;   // 4 groups of 32 rows
  const int waveN = wave >> 2;  // 2 groups of 64 cols

  v8f acc[2][4];
#pragma unroll
  for (int i = 0; i < 2; ++i)
#pragma unroll
    for (int j = 0; j < 4; ++j) acc[i][j] = (v8f){0.f,0.f,0.f,0.f,0.f,0.f,0.f,0.f};

  for (int k0 = 0; k0 < 512; k0 += 32) {
    // Stage 128x32 bf16 tiles of A(hi/lo) and B(hi/lo): 512 16B-chunks each.
#pragma unroll
    for (int p = 0; p < 2; ++p) {
      int c = t + p * 256;           // 0..511
      int r = c >> 2;
      int q = (c & 3) * 8;           // bf16 offset within row, 16B chunks
      size_t ga = (size_t)(rowBase + r) * 512 + k0 + q;
      size_t gb = (size_t)(colBase + r) * 512 + k0 + q;
      int lo_off = r * LD + q;
#if HAVE_ASYNC_LDS
      __builtin_amdgcn_global_load_async_to_lds_b128(
          (AS1 gi4*)(Xhi + ga), (AS3 gi4*)(&AhiS[lo_off]), 0, 0);
      __builtin_amdgcn_global_load_async_to_lds_b128(
          (AS1 gi4*)(Xlo + ga), (AS3 gi4*)(&AloS[lo_off]), 0, 0);
      __builtin_amdgcn_global_load_async_to_lds_b128(
          (AS1 gi4*)(Whi + gb), (AS3 gi4*)(&BhiS[lo_off]), 0, 0);
      __builtin_amdgcn_global_load_async_to_lds_b128(
          (AS1 gi4*)(Wlo + gb), (AS3 gi4*)(&BloS[lo_off]), 0, 0);
#else
      *(u4*)(&AhiS[lo_off]) = *(const u4*)(Xhi + ga);
      *(u4*)(&AloS[lo_off]) = *(const u4*)(Xlo + ga);
      *(u4*)(&BhiS[lo_off]) = *(const u4*)(Whi + gb);
      *(u4*)(&BloS[lo_off]) = *(const u4*)(Wlo + gb);
#endif
    }
#if HAVE_ASYNC_LDS
    __builtin_amdgcn_s_wait_asynccnt(0);   // this wave's LDS writes landed
#endif
    __syncthreads();

    v16bf ah[2], al[2], bh[4], bl[4];
#pragma unroll
    for (int i = 0; i < 2; ++i) {
      int r = (waveM * 32 + i * 16 + sub16) * LD;
      ah[i] = load_a_frag(&AhiS[r], half);
      al[i] = load_a_frag(&AloS[r], half);
    }
#pragma unroll
    for (int j = 0; j < 4; ++j) {
      int r = (waveN * 64 + j * 16 + sub16) * LD;
      bh[j] = load_b_frag(&BhiS[r], half);
      bl[j] = load_b_frag(&BloS[r], half);
    }
#pragma unroll
    for (int i = 0; i < 2; ++i)
#pragma unroll
      for (int j = 0; j < 4; ++j) {
        acc[i][j] = __builtin_amdgcn_wmma_f32_16x16x32_bf16(
            false, ah[i], false, bh[j], (short)0, acc[i][j], false, false);
        acc[i][j] = __builtin_amdgcn_wmma_f32_16x16x32_bf16(
            false, ah[i], false, bl[j], (short)0, acc[i][j], false, false);
        acc[i][j] = __builtin_amdgcn_wmma_f32_16x16x32_bf16(
            false, al[i], false, bh[j], (short)0, acc[i][j], false, false);
      }
    __syncthreads();
  }

  // Store: C/D f32 layout: lane = n + 16*(m>=8); vgpr r -> m = r + 8*(lane>>4)
#pragma unroll
  for (int i = 0; i < 2; ++i)
#pragma unroll
    for (int j = 0; j < 4; ++j) {
      int n = colBase + waveN * 64 + j * 16 + sub16;
      float bias = bqkv[jmap(n)];
#pragma unroll
      for (int r = 0; r < 8; ++r) {
        int m = rowBase + waveM * 32 + i * 16 + half * 8 + r;
        KV[(size_t)m * NKV + n] = acc[i][j][r] + bias;
      }
    }
}

// ---------------------------------------------------------------------------
// Kernel 2: CLS-row Q for both branches (exact fp32): Q[32, 1024] (qw | qE)
// ---------------------------------------------------------------------------
__global__ __launch_bounds__(256) void qcls_kernel(
    const float* __restrict__ X, const float* __restrict__ W,
    const float* __restrict__ bqkv, float* __restrict__ Q)
{
  int c = blockIdx.x * 256 + threadIdx.x;   // 0..1023
  int b = blockIdx.y;
  int j = (c < 512) ? c : (c + 1024);       // qw: [0,512), qE: [1536,2048)
  const float* xr = X + (size_t)b * Ssz * Esz;  // row (b, s=0)
  const float* wr = W + (size_t)j * 512;
  float acc = bqkv[j];
  for (int k = 0; k < 512; k += 4) {
    f4 xv = *(const f4*)(xr + k);
    f4 wv = *(const f4*)(wr + k);
    acc += xv.x * wv.x + xv.y * wv.y + xv.z * wv.z + xv.w * wv.w;
  }
  Q[b * 1024 + c] = acc;
}

// ---------------------------------------------------------------------------
// Kernel 3: single-query attention + residual + LayerNorm + output proj
// grid (B, 2): blockIdx.y = 0 -> w branch, 1 -> E branch. 512 threads.
// ---------------------------------------------------------------------------
__global__ __launch_bounds__(512) void attn_kernel(
    const float* __restrict__ KV, const float* __restrict__ Q,
    const float* __restrict__ X, const int* __restrict__ mask,
    const float* __restrict__ W_ow, const float* __restrict__ b_ow,
    const float* __restrict__ W_oE, const float* __restrict__ b_oE,
    const float* __restrict__ g_w, const float* __restrict__ beta_w,
    const float* __restrict__ g_E, const float* __restrict__ beta_E,
    float* __restrict__ out)
{
  __shared__ float qs[512];
  __shared__ float lg[8 * 512];
  __shared__ float red[512];
  __shared__ float hn[512];

  const int t  = threadIdx.x;
  const int b  = blockIdx.x;
  const int br = blockIdx.y;
  const int koff = br ? 1024 : 0;
  const int voff = koff + 512;
  const float* kvb = KV + (size_t)(b * Ssz) * NKV;

  qs[t] = Q[b * 1024 + br * 512 + t];
  __syncthreads();

  // logits[h][s] = (q_h . k_{s,h}) * sqrt(D)  (reference divides by D^-0.5)
  const int h  = t & 7;
  const int sl = t >> 3;
  const float* qh = qs + h * 64;
  for (int chunk = 0; chunk < 8; ++chunk) {
    int s = chunk * 64 + sl;
    const float* kr = kvb + (size_t)s * NKV + koff + h * 64;
    float dot = 0.f;
#pragma unroll
    for (int d = 0; d < 64; d += 4) {
      f4 kv4 = *(const f4*)(kr + d);
      dot += qh[d] * kv4.x + qh[d+1] * kv4.y + qh[d+2] * kv4.z + qh[d+3] * kv4.w;
    }
    lg[h * 512 + s] = mask[b * Ssz + s] ? dot * 8.0f : -3.0e38f;
  }
  __syncthreads();

  // per-head max (CLS key always valid -> finite max, row_valid always true)
  const int hr = t >> 6;
  const int ir = t & 63;
  float m = -3.0e38f;
#pragma unroll
  for (int jj = 0; jj < 8; ++jj) m = fmaxf(m, lg[hr * 512 + ir + jj * 64]);
  red[t] = m;
  __syncthreads();
  for (int off = 32; off > 0; off >>= 1) {
    if (ir < off) red[t] = fmaxf(red[t], red[t + off]);
    __syncthreads();
  }
  float hmax = red[hr * 64];
  __syncthreads();

  float ssum = 0.f;
#pragma unroll
  for (int jj = 0; jj < 8; ++jj) {
    int idx = hr * 512 + ir + jj * 64;
    float e = __expf(lg[idx] - hmax);
    lg[idx] = e;
    ssum += e;
  }
  red[t] = ssum;
  __syncthreads();
  for (int off = 32; off > 0; off >>= 1) {
    if (ir < off) red[t] += red[t + off];
    __syncthreads();
  }
  float hsum = red[hr * 64];
  __syncthreads();

  // o[t] = (a_h . V[:, t]) / hsum ; residual with x[b, 0, t]
  const float* xr = X + (size_t)(b * Ssz) * Esz;
  const float* aw = lg + hr * 512;
  float acc = 0.f;
  for (int s = 0; s < 512; ++s)
    acc += aw[s] * kvb[(size_t)s * NKV + voff + t];
  float he = acc / hsum + xr[t];
  qs[t] = he;
  __syncthreads();

  // LayerNorm (two-pass)
  red[t] = he;
  __syncthreads();
  for (int off = 256; off > 0; off >>= 1) {
    if (t < off) red[t] += red[t + off];
    __syncthreads();
  }
  float mean = red[0] * (1.0f / 512.0f);
  __syncthreads();
  float d0 = he - mean;
  red[t] = d0 * d0;
  __syncthreads();
  for (int off = 256; off > 0; off >>= 1) {
    if (t < off) red[t] += red[t + off];
    __syncthreads();
  }
  float var = red[0] * (1.0f / 512.0f);
  __syncthreads();
  float rstd = rsqrtf(var + 1e-5f);
  const float* gg = br ? g_E : g_w;
  const float* bb = br ? beta_E : beta_w;
  float hv = d0 * rstd * gg[t] + bb[t];
  hn[t] = hv;
  __syncthreads();

  if (br == 0) {
    red[t] = hv * W_ow[t];
    __syncthreads();
    for (int off = 256; off > 0; off >>= 1) {
      if (t < off) red[t] += red[t + off];
      __syncthreads();
    }
    if (t == 0) out[b] = red[0] + b_ow[0];
  } else {
    const float* wr = W_oE + (size_t)t * 512;
    float a2 = b_oE[t];
    for (int jc = 0; jc < 512; jc += 4) {
      f4 wv = *(const f4*)(wr + jc);
      a2 += hn[jc] * wv.x + hn[jc+1] * wv.y + hn[jc+2] * wv.z + hn[jc+3] * wv.w;
    }
    out[Bsz + b * 512 + t] = a2;
  }
}

extern "C" void kernel_launch(void* const* d_in, const int* in_sizes, int n_in,
                              void* d_out, int out_size, void* d_ws, size_t ws_size,
                              hipStream_t stream) {
  (void)in_sizes; (void)n_in; (void)out_size; (void)ws_size;
  const float* x      = (const float*)d_in[0];
  const float* W_qkv  = (const float*)d_in[1];
  const float* b_qkv  = (const float*)d_in[2];
  const float* W_ow   = (const float*)d_in[3];
  const float* b_ow   = (const float*)d_in[4];
  const float* W_oE   = (const float*)d_in[5];
  const float* b_oE   = (const float*)d_in[6];
  const float* g_w    = (const float*)d_in[7];
  const float* beta_w = (const float*)d_in[8];
  const float* g_E    = (const float*)d_in[9];
  const float* beta_E = (const float*)d_in[10];
  const int*   mask   = (const int*)d_in[11];
  float* out = (float*)d_out;

  // Workspace layout (all offsets 16B aligned)
  char* ws = (char*)d_ws;
  float*  kv   = (float*)ws;                                  // 16384*2048 f32 = 134.2 MB
  float*  qcls = (float*)(ws + (size_t)Mtot * NKV * 4);       // 32*1024 f32
  char*   p    = ws + (size_t)Mtot * NKV * 4 + 32 * 1024 * 4;
  __bf16* Xhi  = (__bf16*)p;                                  // 16384*512 bf16
  __bf16* Xlo  = Xhi + (size_t)Mtot * 512;
  __bf16* Whi  = Xlo + (size_t)Mtot * 512;                    // 2048*512 bf16 (packed rows)
  __bf16* Wlo  = Whi + (size_t)NKV * 512;

  convert_x<<<(Mtot * 512) / (256 * 4), 256, 0, stream>>>(x, Xhi, Xlo);
  convert_w<<<(NKV * 512) / (256 * 4), 256, 0, stream>>>(W_qkv, Whi, Wlo);

  dim3 g1(NKV / 128, Mtot / 128);   // (16, 128)
  qkv_kv_gemm<<<g1, 256, 0, stream>>>(Xhi, Xlo, Whi, Wlo, b_qkv, kv);

  dim3 g2(4, Bsz);
  qcls_kernel<<<g2, 256, 0, stream>>>(x, W_qkv, b_qkv, qcls);

  dim3 g3(Bsz, 2);
  attn_kernel<<<g3, 512, 0, stream>>>(kv, qcls, x, mask, W_ow, b_ow, W_oE, b_oE,
                                      g_w, beta_w, g_E, beta_E, out);
}
